// UnidirectionalLSTMSentenceEncoder_50199577755975
// MI455X (gfx1250) — matrix operand order repaired and checked by reading, not compiled, not run on hardware
//
#include <hip/hip_runtime.h>

// ---------------------------------------------------------------------------
// Unidirectional LSTM sentence encoder for MI455X (gfx1250, wave32, WMMA).
//   B=128, T=64, E=300, H=2048, gates=4H=8192.
//   Per step: gates = h @ W_hh^T (bf16 WMMA, K=2048)
//                   + x @ W_ih^T (bf16 WMMA, K padded 300->320)
//                   + b_ih + b_hh
//   Weights converted once to bf16 in ws (L2-resident across all 64 steps).
// ---------------------------------------------------------------------------

typedef __bf16 bf16_t;
typedef __attribute__((ext_vector_type(16))) __bf16 v16bf;
typedef __attribute__((ext_vector_type(8)))  __bf16 v8bf;
typedef __attribute__((ext_vector_type(8)))  float   v8f;
typedef __attribute__((ext_vector_type(4)))  float   v4f;

#define BATCH 128
#define TLEN  64
#define EDIM  300
#define HDIM  2048
#define G4H   8192   // 4*H

__device__ __forceinline__ float sigm(float x) {
    return 1.0f / (1.0f + __expf(-x));
}
__device__ __forceinline__ float tanh_fast(float x) {
    // robust: x->+inf => 1, x->-inf => -1 (exp saturates to inf / 0)
    return 1.0f - 2.0f / (__expf(2.0f * x) + 1.0f);
}
// guarded float4 load: k is a multiple of 4; rows have E=300 (300 % 4 == 0)
__device__ __forceinline__ v4f ld4g(const float* __restrict__ p, int k) {
    if (k < EDIM) return *(const v4f*)(p + k);
    v4f z = {0.f, 0.f, 0.f, 0.f};
    return z;
}

// --------------------------- fp32 -> bf16 weight convert --------------------
__global__ void convert_whh(const float* __restrict__ w, bf16_t* __restrict__ o) {
    int i = (blockIdx.x * blockDim.x + threadIdx.x) * 8;   // 4H*H = 16,777,216
    v4f a = *(const v4f*)(w + i);
    v4f b = *(const v4f*)(w + i + 4);
    v8bf r;
#pragma unroll
    for (int j = 0; j < 4; ++j) {
        r[j]     = (bf16_t)a[j];
        r[4 + j] = (bf16_t)b[j];
    }
    *(v8bf*)(o + i) = r;
}

// --------------------------- zero h0 / c0 -----------------------------------
__global__ void init_state(bf16_t* __restrict__ h0, float* __restrict__ c0) {
    int i = blockIdx.x * blockDim.x + threadIdx.x;         // B*H = 262,144
    h0[i] = (bf16_t)0.0f;
    c0[i] = 0.0f;
}

// --------------------------- one LSTM timestep ------------------------------
// grid.x = H/16 = 128 column tiles; block = 256 threads = 8 waves,
// wave w owns batch-row tile [16w, 16w+16). Each wave computes a 16x16 D tile
// for each of the 4 gates (A tile shared across gates in registers).
__global__ __launch_bounds__(256, 1) void lstm_step(
    const int*    __restrict__ tokens,   // [B, T]
    const float*  __restrict__ embed,    // [VOCAB, E]
    const float*  __restrict__ w_ih,     // [4H, E]
    const float*  __restrict__ b_ih,     // [4H]
    const float*  __restrict__ b_hh,     // [4H]
    const bf16_t* __restrict__ w_hh_bf,  // [4H, H] bf16
    const bf16_t* __restrict__ h_in,     // [B, H]  bf16
    bf16_t*       __restrict__ h_out,    // [B, H]  bf16
    float*        __restrict__ c_st,     // [B, H]  fp32 (in place)
    float*        __restrict__ out,      // [B, H]  fp32 (d_out)
    int t)
{
    const int tid  = threadIdx.x;
    const int wave = tid >> 5;        // 0..7  -> batch tile
    const int lane = tid & 31;
    const int n    = lane & 15;       // column-within-tile / A-row-within-tile
    const int hi   = lane >> 4;
    const int j0   = blockIdx.x * 16; // column tile base within each gate
    const int col  = j0 + n;
    const int mrow = wave * 16;       // batch row base

    v8f acc[4] = {};                  // fp32 accumulators: i, f, g, o

    // ---------------- recurrent part: h @ W_hh^T  (K = 2048) ----------------
    const bf16_t* arow = h_in + (size_t)(mrow + n) * HDIM;   // A row for this lane
    const bf16_t* bcol[4];
#pragma unroll
    for (int g = 0; g < 4; ++g)       // B column = row (g*H + col) of w_hh
        bcol[g] = w_hh_bf + (size_t)(g * HDIM + col) * HDIM + hi * 16;

    for (int k0 = 0; k0 < HDIM; k0 += 32) {
        // A lane layout (contiguous, matches B): K = k0+hi*16 .. k0+hi*16+15
        v8bf a0 = *(const v8bf*)(arow + k0 + hi * 16);
        v8bf a1 = *(const v8bf*)(arow + k0 + hi * 16 + 8);
        v16bf a;
#pragma unroll
        for (int i = 0; i < 8; ++i) { a[i] = a0[i]; a[8 + i] = a1[i]; }
#pragma unroll
        for (int g = 0; g < 4; ++g) {
            v16bf b = *(const v16bf*)(bcol[g] + k0);  // 16 contiguous K, 32B load
            acc[g] = __builtin_amdgcn_wmma_f32_16x16x32_bf16(
                false, a, false, b, (short)0, acc[g], false, false);
        }
    }

    // ---------------- input part: x @ W_ih^T  (K = 300, padded to 320) ------
    {
        const int   row_b = mrow + n;                      // batch row of A lane
        const int   tok   = tokens[row_b * TLEN + t];
        const float* erow = embed + (size_t)tok * EDIM;
        const float* wrow[4];
#pragma unroll
        for (int g = 0; g < 4; ++g)
            wrow[g] = w_ih + (size_t)(g * HDIM + col) * EDIM;

        for (int k0 = 0; k0 < 320; k0 += 32) {
            const int kb0 = k0 + hi * 16;
            const int kb1 = k0 + hi * 16 + 8;
            v4f f0 = ld4g(erow, kb0), f1 = ld4g(erow, kb0 + 4);
            v4f f2 = ld4g(erow, kb1), f3 = ld4g(erow, kb1 + 4);
            v16bf a;
#pragma unroll
            for (int i = 0; i < 4; ++i) {
                a[i]      = (bf16_t)f0[i];  a[4 + i]  = (bf16_t)f1[i];
                a[8 + i]  = (bf16_t)f2[i];  a[12 + i] = (bf16_t)f3[i];
            }
            const int kB = k0 + hi * 16;
#pragma unroll
            for (int g = 0; g < 4; ++g) {
                v4f g0 = ld4g(wrow[g], kB),     g1 = ld4g(wrow[g], kB + 4);
                v4f g2 = ld4g(wrow[g], kB + 8), g3 = ld4g(wrow[g], kB + 12);
                v16bf b;
#pragma unroll
                for (int i = 0; i < 4; ++i) {
                    b[i]      = (bf16_t)g0[i];  b[4 + i]  = (bf16_t)g1[i];
                    b[8 + i]  = (bf16_t)g2[i];  b[12 + i] = (bf16_t)g3[i];
                }
                acc[g] = __builtin_amdgcn_wmma_f32_16x16x32_bf16(
                    false, a, false, b, (short)0, acc[g], false, false);
            }
        }
    }

    // ---------------- epilogue: biases, nonlinearity, state update ----------
    const float bias_i = b_ih[col]            + b_hh[col];
    const float bias_f = b_ih[HDIM + col]     + b_hh[HDIM + col];
    const float bias_g = b_ih[2 * HDIM + col] + b_hh[2 * HDIM + col];
    const float bias_o = b_ih[3 * HDIM + col] + b_hh[3 * HDIM + col];

#pragma unroll
    for (int i = 0; i < 8; ++i) {
        // D lane layout: element i is row hi*8+i of the tile, column n
        const int    brow = mrow + hi * 8 + i;
        const size_t idx  = (size_t)brow * HDIM + col;
        const float ig = sigm(acc[0][i] + bias_i);
        const float fg = sigm(acc[1][i] + bias_f);
        const float gg = tanh_fast(acc[2][i] + bias_g);
        const float og = sigm(acc[3][i] + bias_o);
        const float cn = fg * c_st[idx] + ig * gg;   // (idx owned by this lane)
        c_st[idx]  = cn;
        const float hn = og * tanh_fast(cn);
        out[idx]   = hn;                             // last step's write survives
        h_out[idx] = (bf16_t)hn;                     // recurrent state, bf16
    }
}

// ---------------------------------------------------------------------------
extern "C" void kernel_launch(void* const* d_in, const int* in_sizes, int n_in,
                              void* d_out, int out_size, void* d_ws, size_t ws_size,
                              hipStream_t stream) {
    const int*   tokens = (const int*)  d_in[0];
    const float* embed  = (const float*)d_in[1];
    const float* w_ih   = (const float*)d_in[2];
    const float* w_hh   = (const float*)d_in[3];
    const float* b_ih   = (const float*)d_in[4];
    const float* b_hh   = (const float*)d_in[5];
    float* out = (float*)d_out;

    // ws layout (total ~34.1 MB):
    //   [0, 33554432)               w_hh in bf16 (4H*H*2)
    //   [33554432, +524288)         h ping buffer (B*H bf16)
    //   [.., +524288)               h pong buffer
    //   [.., +1048576)              c state (B*H fp32)
    char*   ws       = (char*)d_ws;
    bf16_t* w_hh_bf  = (bf16_t*)ws;
    bf16_t* hbuf0    = (bf16_t*)(ws + 33554432);
    bf16_t* hbuf1    = (bf16_t*)(ws + 33554432 + 524288);
    float*  c_st     = (float*) (ws + 33554432 + 1048576);

    // 1) convert W_hh to bf16 (16,777,216 elems, 8 per thread)
    convert_whh<<<(G4H * HDIM / 8) / 256, 256, 0, stream>>>(w_hh, w_hh_bf);
    // 2) zero h0 / c0
    init_state<<<(BATCH * HDIM) / 256, 256, 0, stream>>>(hbuf0, c_st);
    // 3) 64 sequential LSTM steps (stream order carries the dependency)
    for (int t = 0; t < TLEN; ++t) {
        bf16_t* hin  = (t & 1) ? hbuf1 : hbuf0;
        bf16_t* hout = (t & 1) ? hbuf0 : hbuf1;
        lstm_step<<<HDIM / 16, 256, 0, stream>>>(
            tokens, embed, w_ih, b_ih, b_hh, w_hh_bf, hin, hout, c_st, out, t);
    }
}
